// VSTSREncoderTransformerBlock_43241730736493
// MI455X (gfx1250) — matrix operand order repaired and checked by reading, not compiled, not run on hardware
//
#include <hip/hip_runtime.h>
#include <hip/hip_bf16.h>
#include <stdint.h>

typedef unsigned short u16;
typedef unsigned int   u32;
typedef __attribute__((ext_vector_type(16))) __bf16 v16bf;
typedef __attribute__((ext_vector_type(8)))  float  v8f;
typedef int v4i __attribute__((vector_size(16)));
typedef unsigned int u32x4 __attribute__((ext_vector_type(4)));
typedef int i32x8 __attribute__((ext_vector_type(8)));
typedef int i32x4 __attribute__((ext_vector_type(4)));

#define DEV static __device__ __forceinline__

#define NWIN 512
#define NTOK 256
#define CDIM 128
#define HEADS 4
#define HD 32
#define MLPH 512
#define QK_SCALE 0.17677669529663687f   // 32^-0.5

// Native f32 -> bf16 conversion (emits v_cvt_pk_bf16_f32 / v_fma_mix*_bf16).
DEV u16 f2bf(float f){ union{ __bf16 b; u16 s; }c; c.b = (__bf16)f; return c.s; }
DEV __bf16 bfb(u16 s){ union{ u16 s; __bf16 b; }c; c.s=s; return c.b; }

DEV v8f vzero8(){ v8f z;
#pragma unroll
  for(int i=0;i<8;i++) z[i]=0.f; return z; }

DEV v8f wmma_bf16(v16bf a, v16bf b, v8f c){
  return __builtin_amdgcn_wmma_f32_16x16x32_bf16(false,a,false,b,(short)0,c,false,false);
}

// ---- gfx1250 async global->LDS staging (ASYNCcnt path), with safe fallback ----
#if defined(__gfx1250__) && __has_builtin(__builtin_amdgcn_global_load_async_to_lds_b128)
#define HAVE_ASYNC 1
#else
#define HAVE_ASYNC 0
#endif

// ---- gfx1250 Tensor Data Mover (TENSORcnt path) ----
// clang-22 (ROCm 7.2): 5-arg builtin (g0,g1,g2,g3,cpol)
// clang-23 (amdgpu-toolchain): 6-arg builtin (g0,g1,g2,g3,extra,int cpol) -- probe-verified
#if defined(__gfx1250__) && \
    __has_builtin(__builtin_amdgcn_tensor_load_to_lds) && \
    __has_builtin(__builtin_amdgcn_s_wait_tensorcnt)
#define HAVE_TDM 1
#else
#define HAVE_TDM 0
#endif

DEV void cp16(void* l, const void* g){
#if HAVE_ASYNC
  __builtin_amdgcn_global_load_async_to_lds_b128(
      (__attribute__((address_space(1))) v4i*)(void*)g,
      (__attribute__((address_space(3))) v4i*)l, 0, 0);
#else
  *(uint4*)l = *(const uint4*)g;
#endif
}
DEV void cp_wait(){
#if HAVE_ASYNC
#if __has_builtin(__builtin_amdgcn_s_wait_asynccnt)
  __builtin_amdgcn_s_wait_asynccnt(0);
#endif
#endif
}

#if HAVE_TDM
// One-descriptor TDM load of a contiguous region of n dwords (global -> LDS).
// D# built per CDNA5 ISA §8.3/8.4: group0 = {count|flags, lds_addr, global_addr, type=2},
// group1 = {data_size=4B, tensor_dim0=n, tensor_dim1=1, tile_dim0=n, tile_dim1=1,
//           tensor_dim0_stride=n}. Groups 2/3: dims unused.
DEV void tdm_load_1d(u32 lds_addr, const void* g, u32 ndw){
  uint64_t ga = (uint64_t)(uintptr_t)g;
  u32x4 g0; i32x8 g1; i32x4 g2, g3;
  g0[0] = 1u;                                           // count=1 (valid user D#)
  g0[1] = lds_addr;                                     // lds_addr[31:0]
  g0[2] = (u32)ga;                                      // global_addr[31:0]
  g0[3] = (u32)((ga >> 32) & 0x01FFFFFFu) | 0x80000000u;// global_addr[56:32] | type=2
  g1[0] = (int)0x00020000u;                             // data_size=2 (4 bytes)
  g1[1] = (int)((ndw & 0xFFFFu) << 16);                 // tensor_dim0[15:0]
  g1[2] = (int)(((ndw >> 16) & 0xFFFFu) | (1u << 16));  // tensor_dim0[31:16], tensor_dim1=1
  g1[3] = (int)((ndw & 0xFFFFu) << 16);                 // tile_dim0 = ndw
  g1[4] = 1;                                            // tile_dim1=1, tile_dim2=0
  g1[5] = (int)ndw;                                     // tensor_dim0_stride[31:0]
  g1[6] = 0;                                            // stride[47:32], dim1_stride lo
  g1[7] = 0;
  g2[0] = 1; g2[1] = 0; g2[2] = 0; g2[3] = 0;           // tensor_dim2=1, rest unused
  g3[0] = 0; g3[1] = 0; g3[2] = 0; g3[3] = 0;
#if __clang_major__ >= 23
  i32x8 zx;
#pragma unroll
  for(int i=0;i<8;i++) zx[i]=0;                         // trailing slot: zero-filled (null)
  __builtin_amdgcn_tensor_load_to_lds(g0, g1, g2, g3, zx, 0);
#else
  __builtin_amdgcn_tensor_load_to_lds(g0, g1, g2, g3, 0);
#endif
}
#endif

// A fragment (16x32 bf16). p = rowBase + kbase + 16*(lane>>4).
// elems 0..15 -> p[0..15]  (K contiguous per lane, matches B fragments)
DEV v16bf afrag(const u16* p){
  v16bf a;
#pragma unroll
  for(int e=0;e<16;e++) a[e]=bfb(p[e]);
  return a;
}
// B fragment (32x16), K contiguous per lane: k = 16*(lane>>4)+e, col = lane&15
DEV v16bf bfrag_c(const u16* p){
  v16bf b;
#pragma unroll
  for(int e=0;e<16;e++) b[e]=bfb(p[e]);
  return b;
}
// B fragment gathered from f32 weights, element stride = matrix row stride
DEV v16bf bfrag_f32s(const float* p, int stride){
  v16bf b;
#pragma unroll
  for(int e=0;e<16;e++) b[e]=bfb(f2bf(p[e*stride]));
  return b;
}
// B fragment gathered from u16 LDS with element stride
DEV v16bf bfrag_u16s(const u16* p, int stride){
  v16bf b;
#pragma unroll
  for(int e=0;e<16;e++) b[e]=bfb(p[e*stride]);
  return b;
}

// ---------------- Kernel 1: LN1 + shift + window partition + QKV GEMM ----------------
__global__ __launch_bounds__(256) void k_qkv(const float* __restrict__ x,
     const float* __restrict__ g1, const float* __restrict__ b1,
     const float* __restrict__ qw, const float* __restrict__ qb,
     const float* __restrict__ kvw, const float* __restrict__ kvb,
     u16* __restrict__ qo, u16* __restrict__ ko, u16* __restrict__ vo)
{
  __shared__ __align__(16) u16 sX[NTOK*CDIM];   // 64 KB
  const int win = blockIdx.x;
  const int b = win>>8, ih=(win>>4)&15, iw=win&15;
  const int tid=threadIdx.x, lane=tid&31, wid=tid>>5;
  const int hi=lane>>4, ln=lane&15;

  // Phase 1: layernorm + cyclic shift(-4,-4) + partition (one token per wave per pass)
  float gv[4], bv[4];
#pragma unroll
  for(int c=0;c<4;c++){ gv[c]=g1[lane*4+c]; bv[c]=b1[lane*4+c]; }
  for(int t = wid; t < NTOK; t += 8){
    int f = t>>6, wi=(t>>3)&7, wj=t&7;
    int hh = (ih*8+wi+4)&127, ww=(iw*8+wj+4)&127;
    const float* px = x + (size_t)((((b*4+f)*128+hh)*128+ww))*CDIM;
    float v0[4]; float s=0.f, s2=0.f;
#pragma unroll
    for(int c=0;c<4;c++){ float vv = px[lane*4+c]; v0[c]=vv; s+=vv; s2+=vv*vv; }
#pragma unroll
    for(int d=1; d<32; d<<=1){ s += __shfl_xor(s,d); s2 += __shfl_xor(s2,d); }
    float mean = s*(1.f/128.f);
    float var  = s2*(1.f/128.f) - mean*mean;
    float rstd = rsqrtf(var + 1e-5f);
#pragma unroll
    for(int c=0;c<4;c++)
      sX[t*CDIM + lane*4+c] = f2bf((v0[c]-mean)*rstd*gv[c] + bv[c]);
  }
  __syncthreads();

  // Phase 2: GEMM 256x384 = sX(256x128) * [q_w | kv_w]
  for(int nn=0; nn<3; nn++){
    const int nt  = wid*3 + nn;
    const int col = nt*16 + ln;                  // 0..383
    const float* wptr; int wstride;
    if(col < 128){ wptr = qw + col;        wstride = 128; }
    else         { wptr = kvw + (col-128); wstride = 256; }
    __builtin_prefetch(wptr, 0, 3);
    v16bf bf[4];
#pragma unroll
    for(int kt=0;kt<4;kt++) bf[kt] = bfrag_f32s(wptr + (kt*32 + 16*hi)*wstride, wstride);

    const float bias_add = (col<128) ? qb[col] : kvb[col-128];
    const float scale    = (col<128) ? QK_SCALE : 1.f;
    u16* outb; int head, hd;
    if(col<128)      { head=col>>5;          hd=col&31;        outb=qo; }
    else if(col<256) { int cc=col-128; head=cc>>5; hd=cc&31;   outb=ko; }
    else             { int cc=col-256; head=cc>>5; hd=cc&31;   outb=vo; }
    u16* od = outb + ((size_t)(win*HEADS + head)*NTOK)*HD + hd;

    for(int mt=0; mt<16; mt++){
      v8f acc = vzero8();
#pragma unroll
      for(int kt=0;kt<4;kt++){
        v16bf a = afrag(sX + (mt*16+ln)*CDIM + kt*32 + 16*hi);
        acc = wmma_bf16(a, bf[kt], acc);
      }
#pragma unroll
      for(int j=0;j<8;j++){
        int tok = mt*16 + j + 8*hi;
        od[(size_t)tok*HD] = f2bf((acc[j] + bias_add) * scale);
      }
    }
  }
}

// ---------------- Kernel 2: windowed attention per (window, head) ----------------
__global__ __launch_bounds__(256) void k_attn(const u16* __restrict__ qi,
    const u16* __restrict__ ki, const u16* __restrict__ vi,
    const float* __restrict__ relt, const float* __restrict__ maskm,
    u16* __restrict__ ao)
{
  __shared__ __align__(16) u16 sQ[NTOK*HD];     // 16 KB
  __shared__ __align__(16) u16 sK[NTOK*HD];     // 16 KB
  __shared__ __align__(16) u16 sV[NTOK*HD];     // 16 KB
  __shared__ u16 sP[8*16*32];                   // 8 KB per-wave prob staging
  const int wh = blockIdx.x;           // win*4 + head
  const int win = wh>>2, head = wh&3;
  const int tid=threadIdx.x, lane=tid&31, wid=tid>>5;
  const int hi=lane>>4, ln=lane&15;

  {
    const u16* q8 = qi + (size_t)wh*NTOK*HD;
    const u16* k8 = ki + (size_t)wh*NTOK*HD;
    const u16* v8 = vi + (size_t)wh*NTOK*HD;
#if HAVE_TDM
    // Tensor Data Mover: one descriptor per 16KB tile, issued by wave 0 only.
    if(wid == 0){
      tdm_load_1d((u32)(uintptr_t)(__attribute__((address_space(3))) void*)sQ, q8, NTOK*HD/2);
      tdm_load_1d((u32)(uintptr_t)(__attribute__((address_space(3))) void*)sK, k8, NTOK*HD/2);
      tdm_load_1d((u32)(uintptr_t)(__attribute__((address_space(3))) void*)sV, v8, NTOK*HD/2);
      __builtin_amdgcn_s_wait_tensorcnt(0);
    }
#else
#pragma unroll
    for(int i=0;i<4;i++){
      int o = (tid + i*256)*8;              // u16 offset of this 16-byte chunk
      cp16(&sQ[o], q8 + o);
      cp16(&sK[o], k8 + o);
      cp16(&sV[o], v8 + o);
    }
    cp_wait();
#endif
  }
  __syncthreads();

  const float* maskb = maskm + (size_t)(win&255)*65536;

  for(int qq=0; qq<2; qq++){
    const int qt = wid + qq*8;
    // q fragment: K=32 == head_dim, one A frag covers the whole reduction
    v16bf aq = afrag(sQ + (qt*16+ln)*HD + 16*hi);

    v8f S[16];
#pragma unroll
    for(int kt=0;kt<16;kt++){
      v16bf bk = bfrag_c(sK + (kt*16+ln)*HD + 16*hi);
      S[kt] = wmma_bf16(aq, bk, vzero8());
    }
    // relative-position bias (computed arithmetically) + shifted-window mask
#pragma unroll
    for(int kt=0;kt<16;kt++){
      int m = kt*16 + ln;
      int f2 = m>>6, r2=(m>>3)&7, c2=m&7;
#pragma unroll
      for(int j=0;j<8;j++){
        int n = qt*16 + j + 8*hi;
        int f1=n>>6, r1=(n>>3)&7, c1=n&7;
        int idx = (f1-f2+3)*225 + (r1-r2+7)*15 + (c1-c2+7);
        S[kt][j] += relt[idx*HEADS + head] + maskb[n*256 + m];
      }
    }
    // row softmax: each row lives on 16 lanes (one lane-half), element j fixed
    float inv[8];
#pragma unroll
    for(int j=0;j<8;j++){
      float mx = -3.0e38f;
#pragma unroll
      for(int kt=0;kt<16;kt++) mx = fmaxf(mx, S[kt][j]);
#pragma unroll
      for(int d=1; d<16; d<<=1) mx = fmaxf(mx, __shfl_xor(mx, d));
      float sum = 0.f;
#pragma unroll
      for(int kt=0;kt<16;kt++){ float e = __expf(S[kt][j]-mx); S[kt][j]=e; sum += e; }
#pragma unroll
      for(int d=1; d<16; d<<=1) sum += __shfl_xor(sum, d);
      inv[j] = 1.f/sum;
    }
    // attn @ V over 8 key chunks of 32; probs transposed D->A via per-wave LDS slice
    v8f o0 = vzero8(), o1 = vzero8();
    u16* myP = sP + wid*16*32;
#pragma unroll
    for(int c=0;c<8;c++){
#pragma unroll
      for(int j=0;j<8;j++){
        int r = j + 8*hi;
        myP[r*32 + ln]      = f2bf(S[2*c  ][j]*inv[j]);
        myP[r*32 + 16 + ln] = f2bf(S[2*c+1][j]*inv[j]);
      }
      v16bf ap  = afrag(myP + ln*32 + 16*hi);
      v16bf bv0 = bfrag_u16s(sV + (c*32+16*hi)*HD + 0  + ln, HD);
      v16bf bv1 = bfrag_u16s(sV + (c*32+16*hi)*HD + 16 + ln, HD);
      o0 = wmma_bf16(ap, bv0, o0);
      o1 = wmma_bf16(ap, bv1, o1);
    }
#pragma unroll
    for(int j=0;j<8;j++){
      int tok = qt*16 + j + 8*hi;
      u16* po = ao + ((size_t)win*NTOK + tok)*CDIM + head*HD;
      po[ln]    = f2bf(o0[j]);
      po[16+ln] = f2bf(o1[j]);
    }
  }
}

// ---------------- Kernel 3: proj GEMM + window reverse + un-roll + residual ----------------
__global__ __launch_bounds__(256) void k_proj(const u16* __restrict__ ai,
    const float* __restrict__ pw, const float* __restrict__ pb,
    const float* __restrict__ x, float* __restrict__ y)
{
  __shared__ __align__(16) u16 sA[NTOK*CDIM];   // 64 KB
  const int win = blockIdx.x;
  const int b = win>>8, ih=(win>>4)&15, iw=win&15;
  const int tid=threadIdx.x, lane=tid&31, wid=tid>>5;
  const int hi=lane>>4, ln=lane&15;
  {
    const u16* s8 = ai + (size_t)win*NTOK*CDIM;
#pragma unroll
    for(int i=0;i<16;i++){
      int o = (tid + i*256)*8;
      cp16(&sA[o], s8 + o);
    }
    cp_wait();
  }
  __syncthreads();

  const int col = wid*16 + ln;                  // 0..127
  __builtin_prefetch(pw + col, 0, 3);
  v16bf bf[4];
#pragma unroll
  for(int kt=0;kt<4;kt++) bf[kt] = bfrag_f32s(pw + (kt*32+16*hi)*CDIM + col, CDIM);
  const float bias = pb[col];

  for(int mt=0; mt<16; mt++){
    v8f acc = vzero8();
#pragma unroll
    for(int kt=0;kt<4;kt++){
      v16bf a = afrag(sA + (mt*16+ln)*CDIM + kt*32 + 16*hi);
      acc = wmma_bf16(a, bf[kt], acc);
    }
#pragma unroll
    for(int j=0;j<8;j++){
      int tok = mt*16 + j + 8*hi;
      int f = tok>>6, wi=(tok>>3)&7, wj=tok&7;
      int hh=(ih*8+wi+4)&127, ww=(iw*8+wj+4)&127;
      size_t gi = ((((size_t)(b*4+f)*128+hh)*128+ww))*CDIM + col;
      y[gi] = x[gi] + acc[j] + bias;
    }
  }
}

// ---------------- Kernel 4: LN2 + fc1 GEMM + exact-erf GELU ----------------
__global__ __launch_bounds__(256) void k_fc1(const float* __restrict__ y,
    const float* __restrict__ g2, const float* __restrict__ b2,
    const float* __restrict__ w1, const float* __restrict__ bb1,
    u16* __restrict__ hid)
{
  __shared__ __align__(16) u16 sX[NTOK*CDIM];   // 64 KB
  const int t0 = blockIdx.x * NTOK;
  const int tid=threadIdx.x, lane=tid&31, wid=tid>>5;
  const int hi=lane>>4, ln=lane&15;

  float gv[4], bv[4];
#pragma unroll
  for(int c=0;c<4;c++){ gv[c]=g2[lane*4+c]; bv[c]=b2[lane*4+c]; }
  for(int t=wid; t<NTOK; t+=8){
    const float* py = y + (size_t)(t0+t)*CDIM;
    float v0[4]; float s=0.f, s2=0.f;
#pragma unroll
    for(int c=0;c<4;c++){ float vv=py[lane*4+c]; v0[c]=vv; s+=vv; s2+=vv*vv; }
#pragma unroll
    for(int d=1;d<32;d<<=1){ s+=__shfl_xor(s,d); s2+=__shfl_xor(s2,d); }
    float mean=s*(1.f/128.f), var=s2*(1.f/128.f)-mean*mean, rstd=rsqrtf(var+1e-5f);
#pragma unroll
    for(int c=0;c<4;c++) sX[t*CDIM+lane*4+c]=f2bf((v0[c]-mean)*rstd*gv[c]+bv[c]);
  }
  __syncthreads();

  for(int nn=0;nn<4;nn++){
    const int col = (wid*4+nn)*16 + ln;         // 0..511
    __builtin_prefetch(w1 + col, 0, 3);
    v16bf bf[4];
#pragma unroll
    for(int kt=0;kt<4;kt++) bf[kt] = bfrag_f32s(w1 + (kt*32+16*hi)*MLPH + col, MLPH);
    const float bias = bb1[col];
    for(int mt=0;mt<16;mt++){
      v8f acc = vzero8();
#pragma unroll
      for(int kt=0;kt<4;kt++){
        v16bf a = afrag(sX + (mt*16+ln)*CDIM + kt*32 + 16*hi);
        acc = wmma_bf16(a, bf[kt], acc);
      }
#pragma unroll
      for(int j=0;j<8;j++){
        int tok = mt*16 + j + 8*hi;
        float h = acc[j] + bias;
        float g = 0.5f*h*(1.f + erff(h*0.70710678118654752f));
        hid[(size_t)(t0+tok)*MLPH + col] = f2bf(g);
      }
    }
  }
}

// ---------------- Kernel 5: fc2 GEMM (K=512 in 4 LDS chunks) + residual ----------------
__global__ __launch_bounds__(256) void k_fc2(const u16* __restrict__ hid,
    const float* __restrict__ w2, const float* __restrict__ bb2,
    const float* __restrict__ y, float* __restrict__ out)
{
  __shared__ __align__(16) u16 sA[NTOK*CDIM];   // 64 KB
  const int t0 = blockIdx.x * NTOK;
  const int tid=threadIdx.x, lane=tid&31, wid=tid>>5;
  const int hi=lane>>4, ln=lane&15;
  const int col = wid*16 + ln;                  // 0..127

  v8f acc[16];
#pragma unroll
  for(int mt=0;mt<16;mt++) acc[mt] = vzero8();

  for(int kc=0;kc<4;kc++){
    __syncthreads();
    {
      const u16* src = hid + (size_t)(t0+tid)*MLPH + kc*CDIM;
      u16* dst = sA + tid*CDIM;
#pragma unroll
      for(int i=0;i<16;i++) cp16(dst + i*8, src + i*8);
      cp_wait();
    }
    __syncthreads();
    __builtin_prefetch(w2 + (kc*CDIM)*CDIM + col, 0, 3);
    v16bf bf[4];
#pragma unroll
    for(int kt=0;kt<4;kt++) bf[kt] = bfrag_f32s(w2 + (kc*CDIM + kt*32 + 16*hi)*CDIM + col, CDIM);
#pragma unroll
    for(int mt=0;mt<16;mt++){
#pragma unroll
      for(int kt=0;kt<4;kt++){
        v16bf a = afrag(sA + (mt*16+ln)*CDIM + kt*32 + 16*hi);
        acc[mt] = wmma_bf16(a, bf[kt], acc[mt]);
      }
    }
  }
  const float bias = bb2[col];
#pragma unroll
  for(int mt=0;mt<16;mt++){
#pragma unroll
    for(int j=0;j<8;j++){
      int tok = mt*16 + j + 8*hi;
      size_t gi = (size_t)(t0+tok)*CDIM + col;
      out[gi] = acc[mt][j] + bias + y[gi];
    }
  }
}

// ---------------- host launcher ----------------
extern "C" void kernel_launch(void* const* d_in, const int* in_sizes, int n_in,
                              void* d_out, int out_size, void* d_ws, size_t ws_size,
                              hipStream_t stream) {
  const float* x      = (const float*)d_in[0];
  const float* maskm  = (const float*)d_in[1];
  const float* n1g    = (const float*)d_in[2];
  const float* n1b    = (const float*)d_in[3];
  const float* qw     = (const float*)d_in[4];
  const float* qb     = (const float*)d_in[5];
  const float* kvw    = (const float*)d_in[6];
  const float* kvb    = (const float*)d_in[7];
  const float* pw     = (const float*)d_in[8];
  const float* pb     = (const float*)d_in[9];
  const float* relt   = (const float*)d_in[10];
  const float* n2g    = (const float*)d_in[11];
  const float* n2b    = (const float*)d_in[12];
  const float* w1     = (const float*)d_in[13];
  const float* bb1    = (const float*)d_in[14];
  const float* w2     = (const float*)d_in[15];
  const float* bb2    = (const float*)d_in[16];

  char* ws = (char*)d_ws;
  // bf16 tensors, 512*256*128*2 = 33554432 bytes each
  u16*  qws = (u16*)(ws + 0);
  u16*  kws = (u16*)(ws + (size_t)33554432);
  u16*  vws = (u16*)(ws + (size_t)67108864);
  u16*  aow = (u16*)(ws + (size_t)100663296);
  float* yw = (float*)(ws + (size_t)134217728);     // 67108864 bytes f32
  u16*  hw  = (u16*)(ws + 0);                       // hidden reuses q/k/v/ao region (134217728 B)

  k_qkv <<<NWIN,        256, 0, stream>>>(x, n1g, n1b, qw, qb, kvw, kvb, qws, kws, vws);
  k_attn<<<NWIN*HEADS,  256, 0, stream>>>(qws, kws, vws, relt, maskm, aow);
  k_proj<<<NWIN,        256, 0, stream>>>(aow, pw, pb, x, yw);
  k_fc1 <<<512,         256, 0, stream>>>(yw, n2g, n2b, w1, bb1, hw);
  k_fc2 <<<512,         256, 0, stream>>>(hw, w2, bb2, yw, (float*)d_out);
}
